// Attention_71820443124456
// MI455X (gfx1250) — matrix operand order repaired and checked by reading, not compiled, not run on hardware
//
#include <hip/hip_runtime.h>
#include <hip/hip_bf16.h>
#include <math.h>

// ---------------- problem constants ----------------
constexpr int B_   = 32;
constexpr int T_   = 1536;
constexpr int H_   = 1024;
constexpr int CTX_ = 1024;
constexpr int KC_  = 32;          // conv_out

// main-kernel tiling
constexpr int MT_  = 32;          // rows of T per workgroup
constexpr int LDA_ = H_ + 8;      // padded LDS stride (bf16 elems) -> conflict-free frag loads
constexpr int LDL_ = KC_ + 8;     // padded loc-tile stride
constexpr int TS_  = 12;          // T-chunks for the context GEMV (384 WGs stream 201 MB)

typedef __attribute__((ext_vector_type(16))) __bf16 v16bf;
typedef __attribute__((ext_vector_type(8)))  __bf16 v8bf;
typedef __attribute__((ext_vector_type(4)))  __bf16 v4bf;
typedef __attribute__((ext_vector_type(8)))  float  f32x8;

// Branch-free tanh: native V_TANH_F32 on CDNA5 if the builtin exists, else exp-based.
__device__ __forceinline__ float fast_tanh(float x) {
#if __has_builtin(__builtin_amdgcn_tanhf)
    return __builtin_amdgcn_tanhf(x);
#else
    float xc = fminf(10.f, fmaxf(-10.f, x));
    float e2 = __expf(2.f * xc);
    return (e2 - 1.f) / (e2 + 1.f);
#endif
}

// ---------------- pre-pass kernels ----------------

// Convert V (CTX,H) and U (CTX,KC) to bf16 (grid-stride).
__global__ void k_cvt_bf16(const float* __restrict__ V, const float* __restrict__ U,
                           __bf16* __restrict__ Vb, __bf16* __restrict__ Ub) {
    int stride = gridDim.x * blockDim.x;
    int i0 = blockIdx.x * blockDim.x + threadIdx.x;
    for (int i = i0; i < CTX_ * H_; i += stride) Vb[i] = (__bf16)V[i];
    for (int i = i0; i < CTX_ * KC_; i += stride) Ub[i] = (__bf16)U[i];
}

// Location conv: loc[b,t,k] = sum_d conv_w[k,0,d]*align[b,t+d-1] + conv_b[k]  (bf16 out)
__global__ void k_conv(const float* __restrict__ al, const float* __restrict__ cw,
                       const float* __restrict__ cb, __bf16* __restrict__ loc) {
    int idx = blockIdx.x * blockDim.x + threadIdx.x;     // over B*T
    if (idx >= B_ * T_) return;
    int b = idx / T_, t = idx - b * T_;
    float a0 = (t > 0)      ? al[b * T_ + t - 1] : 0.f;
    float a1 =                al[b * T_ + t];
    float a2 = (t < T_ - 1) ? al[b * T_ + t + 1] : 0.f;
    __bf16* o = loc + (size_t)idx * KC_;
#pragma unroll
    for (int k = 0; k < KC_; ++k) {
        float v = cw[k * 3 + 0] * a0 + cw[k * 3 + 1] * a1 + cw[k * 3 + 2] * a2 + cb[k];
        o[k] = (__bf16)v;
    }
}

// dec_proj+bias: dpb[b,c] = sum_h dec[b,h]*W[c,h] + bias[c]  (tiny f32 GEMM, float4 dots)
__global__ void k_decproj(const float* __restrict__ dec, const float* __restrict__ W,
                          const float* __restrict__ bias, float* __restrict__ dpb) {
    int c = blockIdx.x * blockDim.x + threadIdx.x;       // grid (CTX/256, B)
    int b = blockIdx.y;
    const float4* d4 = (const float4*)(dec + (size_t)b * H_);
    const float4* w4 = (const float4*)(W + (size_t)c * H_);
    float s = 0.f;
#pragma unroll 4
    for (int i = 0; i < H_ / 4; ++i) {
        float4 a = d4[i], w = w4[i];
        s += a.x * w.x + a.y * w.y + a.z * w.z + a.w * w.w;
    }
    dpb[b * CTX_ + c] = s + bias[c];
}

// ---------------- fused energy/score kernel (WMMA) ----------------
// One WG = batch b, 32 rows of T. 8 waves = 2 (M sub-tiles of 16) x 4 (N stripes of 16).
// For each 64-wide c-stripe: 32x v_wmma_f32_16x16x32_bf16 over K=H, +1 WMMA for the
// location term (K=KC=32), then tanh + w[c] folded into per-row score accumulators.
__global__ __launch_bounds__(256) void
k_scores(const float* __restrict__ enc, const __bf16* __restrict__ Vb,
         const __bf16* __restrict__ Ub, const __bf16* __restrict__ locb,
         const float* __restrict__ dpb, const float* __restrict__ wv,
         float* __restrict__ scores) {
    extern __shared__ __align__(16) char smem_raw[];
    __bf16* As  = (__bf16*)smem_raw;                       // [MT_][LDA_] enc tile (bf16)
    __bf16* Lc  = As + MT_ * LDA_;                         // [MT_][LDL_] loc tile
    float*  Sc  = (float*)(Lc + MT_ * LDL_);               // [MT_] score accumulators

    const int tid  = threadIdx.x;
    const int b    = blockIdx.y;
    const int t0   = blockIdx.x * MT_;
    const int lane = tid & 31;
    const int wave = tid >> 5;
    const int mi   = (wave & 1) * 16;                      // M sub-tile
    const int ni   = wave >> 1;                            // N stripe within 64-chunk

    // ---- stage enc tile (f32 -> bf16) into LDS ----
    {
        const float4* e4 = (const float4*)(enc + ((size_t)b * T_ + t0) * H_);
        for (int i = tid; i < MT_ * H_ / 4; i += 256) {
            int row = (i * 4) / H_, col = (i * 4) % H_;
            float4 f = e4[i];
            v4bf p = { (__bf16)f.x, (__bf16)f.y, (__bf16)f.z, (__bf16)f.w };
            *(v4bf*)(As + row * LDA_ + col) = p;
        }
        const v4bf* l4 = (const v4bf*)(locb + ((size_t)b * T_ + t0) * KC_);
        for (int i = tid; i < MT_ * KC_ / 4; i += 256) {
            int row = (i * 4) / KC_, col = (i * 4) % KC_;
            *(v4bf*)(Lc + row * LDL_ + col) = l4[i];
        }
        if (tid < MT_) Sc[tid] = 0.f;
    }
    __syncthreads();

    // fragment addressing per ISA 7.12.2 (16-bit A 16x32; B 32x16)
    const int koffA = (lane & 16) ? 8 : 0;                 // A: [koffA,+8) U [koffA+16,+8)
    const int koffB = (lane & 16) ? 8 : 0;                 // B: [koffB,+8) U [koffB+16,+8)
    const __bf16* Arow = As + (mi + (lane & 15)) * LDA_ + koffA;

    float sacc[8];
#pragma unroll
    for (int r = 0; r < 8; ++r) sacc[r] = 0.f;

    for (int cc = 0; cc < CTX_ / 64; ++cc) {
        const int c = cc * 64 + ni * 16 + (lane & 15);
        const __bf16* Brow = Vb + (size_t)c * H_ + koffB;
        if (cc + 1 < CTX_ / 64)                            // pull next c-stripe of V toward L2/L0
            __builtin_prefetch((const void*)(Brow + (size_t)64 * H_), 0, 1);

        f32x8 acc = {};
#pragma unroll 8
        for (int k = 0; k < H_; k += 32) {
            v8bf alo = *(const v8bf*)(Arow + k);
            v8bf ahi = *(const v8bf*)(Arow + k + 16);
            v16bf a  = __builtin_shufflevector(alo, ahi, 0,1,2,3,4,5,6,7,8,9,10,11,12,13,14,15);
            v8bf blo = *(const v8bf*)(Brow + k);
            v8bf bhi = *(const v8bf*)(Brow + k + 16);
            v16bf bm = __builtin_shufflevector(blo, bhi, 0,1,2,3,4,5,6,7,8,9,10,11,12,13,14,15);
            acc = __builtin_amdgcn_wmma_f32_16x16x32_bf16(false, a, false, bm,
                                                          (short)0, acc, false, false);
        }
        // location term: one K=32 WMMA against U
        {
            const __bf16* Lrow = Lc + (mi + (lane & 15)) * LDL_ + koffA;
            v8bf alo = *(const v8bf*)(Lrow);
            v8bf ahi = *(const v8bf*)(Lrow + 16);
            v16bf a  = __builtin_shufflevector(alo, ahi, 0,1,2,3,4,5,6,7,8,9,10,11,12,13,14,15);
            const __bf16* Urow = Ub + (size_t)c * KC_ + koffB;
            v8bf blo = *(const v8bf*)(Urow);
            v8bf bhi = *(const v8bf*)(Urow + 16);
            v16bf bm = __builtin_shufflevector(blo, bhi, 0,1,2,3,4,5,6,7,8,9,10,11,12,13,14,15);
            acc = __builtin_amdgcn_wmma_f32_16x16x32_bf16(false, a, false, bm,
                                                          (short)0, acc, false, false);
        }
        const float db = dpb[b * CTX_ + c];                // dec_proj + bias, per column c
        const float wc = wv[c];
#pragma unroll
        for (int r = 0; r < 8; ++r)
            sacc[r] += wc * fast_tanh(acc[r] + db);        // row m = mi + r + (lane>=16 ? 8:0)
    }

    // reduce the 16 columns held across each half-wave, then combine across waves in LDS
#pragma unroll
    for (int r = 0; r < 8; ++r) {
        float v = sacc[r];
        v += __shfl_xor(v, 8, 32);
        v += __shfl_xor(v, 4, 32);
        v += __shfl_xor(v, 2, 32);
        v += __shfl_xor(v, 1, 32);
        if ((lane & 15) == 0)
            atomicAdd(&Sc[mi + r + ((lane & 16) ? 8 : 0)], v);
    }
    __syncthreads();
    if (tid < MT_) scores[b * T_ + t0 + tid] = Sc[tid];
}

// ---------------- softmax over T per batch ----------------
__global__ void k_softmax(const float* __restrict__ scores, float* __restrict__ align) {
    int b = blockIdx.x, tid = threadIdx.x;
    __shared__ float red[256];
    float m = -INFINITY;
    for (int t = tid; t < T_; t += 256) m = fmaxf(m, scores[b * T_ + t]);
    red[tid] = m; __syncthreads();
    for (int s = 128; s > 0; s >>= 1) { if (tid < s) red[tid] = fmaxf(red[tid], red[tid + s]); __syncthreads(); }
    m = red[0]; __syncthreads();
    float sum = 0.f;
    for (int t = tid; t < T_; t += 256) {
        float e = __expf(scores[b * T_ + t] - m);
        align[b * T_ + t] = e;
        sum += e;
    }
    red[tid] = sum; __syncthreads();
    for (int s = 128; s > 0; s >>= 1) { if (tid < s) red[tid] += red[tid + s]; __syncthreads(); }
    float inv = 1.f / red[0];
    for (int t = tid; t < T_; t += 256) align[b * T_ + t] *= inv;
}

// ---------------- context GEMV, two stages for HBM saturation ----------------
// Stage 1: partial[ts][b][h] = sum_{t in chunk ts} align[b,t]*enc[b,t,h]
__global__ void k_context_partial(const float* __restrict__ enc, const float* __restrict__ align,
                                  float* __restrict__ partial) {
    int ts = blockIdx.x, b = blockIdx.y, tid = threadIdx.x;   // 256 threads * float4 = H
    const int tlen = T_ / TS_;
    const int tstart = ts * tlen;
    const float4* e4 = (const float4*)(enc + (size_t)b * T_ * H_);
    float4 acc = {0.f, 0.f, 0.f, 0.f};
    for (int t = tstart; t < tstart + tlen; ++t) {
        float a = align[b * T_ + t];
        float4 v = e4[(size_t)t * (H_ / 4) + tid];
        acc.x += a * v.x; acc.y += a * v.y; acc.z += a * v.z; acc.w += a * v.w;
    }
    ((float4*)partial)[((size_t)ts * B_ + b) * (H_ / 4) + tid] = acc;
}

// Stage 2: ctx[b,h] = sum_ts partial[ts][b][h]
__global__ void k_context_reduce(const float* __restrict__ partial, float* __restrict__ ctx) {
    int i = blockIdx.x * blockDim.x + threadIdx.x;            // over B*H/4 float4s
    if (i >= B_ * H_ / 4) return;
    float4 acc = {0.f, 0.f, 0.f, 0.f};
#pragma unroll
    for (int ts = 0; ts < TS_; ++ts) {
        float4 v = ((const float4*)partial)[(size_t)ts * (B_ * H_ / 4) + i];
        acc.x += v.x; acc.y += v.y; acc.z += v.z; acc.w += v.w;
    }
    ((float4*)ctx)[i] = acc;
}

// ---------------- host launch ----------------
extern "C" void kernel_launch(void* const* d_in, const int* in_sizes, int n_in,
                              void* d_out, int out_size, void* d_ws, size_t ws_size,
                              hipStream_t stream) {
    const float* dec  = (const float*)d_in[0];   // (B,1,H)
    const float* enc  = (const float*)d_in[1];   // (B,T,H)
    const float* lal  = (const float*)d_in[2];   // (B,T)
    const float* W    = (const float*)d_in[3];   // (CTX,H)
    const float* V    = (const float*)d_in[4];   // (CTX,H)
    const float* U    = (const float*)d_in[5];   // (CTX,KC)
    const float* bia  = (const float*)d_in[6];   // (CTX,)
    const float* wv   = (const float*)d_in[7];   // (CTX,)
    const float* cw   = (const float*)d_in[8];   // (KC,1,3)
    const float* cb   = (const float*)d_in[9];   // (KC,)

    float* out_ctx   = (float*)d_out;            // (B,H)
    float* out_align = (float*)d_out + B_ * H_;  // (B,T)

    // workspace carve-up (256B aligned slabs)
    char* ws = (char*)d_ws;
    float*  dpb     = (float*)ws;                         ws += ((size_t)B_ * CTX_ * 4 + 255) & ~255ull;
    float*  scores  = (float*)ws;                         ws += ((size_t)B_ * T_ * 4 + 255) & ~255ull;
    __bf16* Vb      = (__bf16*)ws;                        ws += ((size_t)CTX_ * H_ * 2 + 255) & ~255ull;
    __bf16* Ub      = (__bf16*)ws;                        ws += ((size_t)CTX_ * KC_ * 2 + 255) & ~255ull;
    __bf16* locb    = (__bf16*)ws;                        ws += ((size_t)B_ * T_ * KC_ * 2 + 255) & ~255ull;
    float*  partial = (float*)ws;                         // TS_ * B_ * H_ floats (1.5 MB)

    k_cvt_bf16<<<1024, 256, 0, stream>>>(V, U, Vb, Ub);
    k_conv<<<(B_ * T_ + 255) / 256, 256, 0, stream>>>(lal, cw, cb, locb);
    k_decproj<<<dim3(CTX_ / 256, B_), 256, 0, stream>>>(dec, W, bia, dpb);

    size_t smem = (size_t)(MT_ * LDA_ + MT_ * LDL_) * sizeof(__bf16) + MT_ * sizeof(float);
    k_scores<<<dim3(T_ / MT_, B_), 256, smem, stream>>>(enc, Vb, Ub, locb, dpb, wv, scores);

    k_softmax<<<B_, 256, 0, stream>>>(scores, out_align);
    k_context_partial<<<dim3(TS_, B_), 256, 0, stream>>>(enc, out_align, partial);
    k_context_reduce<<<(B_ * H_ / 4 + 255) / 256, 256, 0, stream>>>(partial, out_ctx);
}
